// CrossViewConLoss_66709432042025
// MI455X (gfx1250) — matrix operand order repaired and checked by reading, not compiled, not run on hardware
//
#include <hip/hip_runtime.h>
#include <hip/hip_bf16.h>
#include <math.h>

typedef __attribute__((ext_vector_type(16))) _Float16 v16h;
typedef __attribute__((ext_vector_type(8)))  _Float16 v8h;
typedef __attribute__((ext_vector_type(8)))  float    v8f;
typedef __attribute__((ext_vector_type(4)))  int      v4i;

#define BATCH_N 2048
#define VIEWS   3
#define DIM     512
#define NROWS   (BATCH_N * VIEWS)        // 6144

#define LDS_STRIDE   520                 // 512 + 8 halves pad -> bank rotation
#define YSPLIT       8
#define ROWS_PER_BLK 128                 // 8 waves x 16 rows
#define COLS_PER_BLK (NROWS / YSPLIT)    // 768
#define JSTEPS       (COLS_PER_BLK / 16) // 48
#define BUF_HALVES   (16 * LDS_STRIDE)

// ---- CDNA5 async global->LDS path (guarded: falls back to load+ds_store) ----
#if defined(__has_builtin)
#if __has_builtin(__builtin_amdgcn_global_load_async_to_lds_b128)
#define HAVE_ASYNC_LDS 1
#endif
#endif
#ifndef HAVE_ASYNC_LDS
#define HAVE_ASYNC_LDS 0
#endif

#if HAVE_ASYNC_LDS
typedef __attribute__((address_space(1))) v4i as1_v4i;   // global v4i
typedef __attribute__((address_space(3))) v4i as3_v4i;   // LDS v4i
#if __has_builtin(__builtin_amdgcn_s_wait_asynccnt)
#define WAIT_ASYNC(n) __builtin_amdgcn_s_wait_asynccnt(n)
#else
#define WAIT_ASYNC(n) asm volatile("s_wait_asynccnt %0" ::"i"(n) : "memory")
#endif
#else
#define WAIT_ASYNC(n)
#endif

// ---------------------------------------------------------------------------
// Kernel 1: row L2-normalize into fp16 (permuted row r = view*BATCH + b reads
// features[b, view, :]) and zero the per-row num/den accumulators.
// ---------------------------------------------------------------------------
__global__ __launch_bounds__(256) void simloss_norm(const float* __restrict__ feat,
                                                    _Float16* __restrict__ fnh,
                                                    float* __restrict__ num,
                                                    float* __restrict__ den) {
    const int r = blockIdx.x;
    const int v = r / BATCH_N;
    const int b = r - v * BATCH_N;
    const float* src = feat + ((size_t)b * VIEWS + v) * DIM;

    float vals[2];
    float ss = 0.f;
#pragma unroll
    for (int t = 0; t < 2; ++t) {
        vals[t] = src[threadIdx.x + t * 256];
        ss += vals[t] * vals[t];
    }
#pragma unroll
    for (int m = 1; m < 32; m <<= 1) ss += __shfl_xor(ss, m, 32);

    __shared__ float red[8];
    if ((threadIdx.x & 31) == 0) red[threadIdx.x >> 5] = ss;
    __syncthreads();
    float tot = red[0];
#pragma unroll
    for (int w = 1; w < 8; ++w) tot += red[w];

    const float scale = 1.0f / fmaxf(sqrtf(tot), 1e-8f);
#pragma unroll
    for (int t = 0; t < 2; ++t)
        fnh[(size_t)r * DIM + threadIdx.x + t * 256] = (_Float16)(vals[t] * scale);

    if (threadIdx.x == 0) { num[r] = 0.f; den[r] = 0.f; }
}

// ---------------------------------------------------------------------------
// B-tile staging: 16 columns x 512 halves -> LDS (padded stride), 16B chunks,
// 4 chunks per thread. Async (ASYNCcnt) when available.
// ---------------------------------------------------------------------------
__device__ __forceinline__ void stage_btile(_Float16* lbuf,
                                            const _Float16* __restrict__ fnh,
                                            int jtile, int tid) {
#pragma unroll
    for (int q = 0; q < 4; ++q) {
        const int t   = tid + q * 256;
        const int col = t >> 6;              // 64 chunks of 8 halves per column
        const int c8  = t & 63;
        const _Float16* g = fnh + ((size_t)(jtile + col)) * DIM + c8 * 8;
        _Float16* l = lbuf + col * LDS_STRIDE + c8 * 8;
#if HAVE_ASYNC_LDS
        __builtin_amdgcn_global_load_async_to_lds_b128((as1_v4i*)g, (as3_v4i*)l, 0, 0);
#else
        *(v8h*)l = *(const v8h*)g;
#endif
    }
}

// ---------------------------------------------------------------------------
// Kernel 2: fused sim GEMM + exp epilogue.
// grid = (NROWS/128, 8), block = 256 (8 waves).
// Wave w keeps its 16-row A tile in VGPRs (16 K-fragments = 128 VGPRs);
// all waves share a double-buffered 16-column B tile staged in LDS.
// ---------------------------------------------------------------------------
__global__ __launch_bounds__(256) void simloss_gemm(const _Float16* __restrict__ fnh,
                                                    float* __restrict__ num,
                                                    float* __restrict__ den) {
    __shared__ __align__(16) _Float16 Btile[2 * BUF_HALVES];   // 2 x 16.6 KB

    const int tid  = threadIdx.x;
    const int lane = tid & 31;
    const int wave = tid >> 5;
    const int half = lane >> 4;
    const int l15  = lane & 15;

    const int ibase = blockIdx.x * ROWS_PER_BLK;
    const int wrow  = ibase + wave * 16;
    const int j0    = blockIdx.y * COLS_PER_BLK;

    // Preload this wave's A tile (v_wmma f16 A layout: M = l15,
    // h[0..7] = K[kb..kb+7], h[8..15] = K[kb+16..kb+23], kb = half*8).
    const _Float16* arow = fnh + ((size_t)(wrow + l15)) * DIM + half * 8;
    v16h afrag[16];
#pragma unroll
    for (int f = 0; f < 16; ++f) {
        const int kk = f * 32;
        v8h alo = *(const v8h*)(arow + kk);
        v8h ahi = *(const v8h*)(arow + kk + 16);
#pragma unroll
        for (int t = 0; t < 8; ++t) { afrag[f][t] = alo[t]; afrag[f][t + 8] = ahi[t]; }
    }

    float numacc[8], denacc[8];
#pragma unroll
    for (int r = 0; r < 8; ++r) { numacc[r] = 0.f; denacc[r] = 0.f; }

    const int iblk = ibase / BATCH_N;    // 2048 % 128 == 0 -> uniform per block

    stage_btile(Btile, fnh, j0, tid);    // prologue -> buffer 0

    for (int js = 0; js < JSTEPS; ++js) {
        _Float16* cur = Btile + (js & 1) * BUF_HALVES;
        if (js + 1 < JSTEPS) {
            // Issue next tile, then wait until only those 4 remain in flight.
            stage_btile(Btile + ((js + 1) & 1) * BUF_HALVES, fnh,
                        j0 + (js + 1) * 16, tid);
            WAIT_ASYNC(4);
        } else {
            WAIT_ASYNC(0);
        }
        __syncthreads();                 // cur tile visible to all waves

        const int jj = j0 + js * 16;
        // B layout: N = l15, h[0..7] = K[kb..kb+7], h[8..15] = K[kb+16..kb+23], kb = half*8.
        const _Float16* brow = cur + l15 * LDS_STRIDE + half * 8;

        v8f c = {};
#pragma unroll
        for (int f = 0; f < 16; ++f) {
            const int kk = f * 32;
            v8h blo = *(const v8h*)(brow + kk);
            v8h bhi = *(const v8h*)(brow + kk + 16);
            v16h bfr;
#pragma unroll
            for (int t = 0; t < 8; ++t) { bfr[t] = blo[t]; bfr[t + 8] = bhi[t]; }
            c = __builtin_amdgcn_wmma_f32_16x16x32_f16(
                    false, afrag[f], false, bfr, (short)0, c, false, false);
        }

        // Block-diagonal mask is tile-uniform (origins are multiples of 16).
        const float m = (iblk == (jj / BATCH_N)) ? 1.0f : 0.0f;
#pragma unroll
        for (int r = 0; r < 8; ++r) {
            const float s = c[r];        // row wrow + r + 8*half, col jj + l15
            denacc[r] += __expf(fabsf(s));
            numacc[r] += m * __expf(s);
        }
        __syncthreads();                 // protect cur before it is restaged
    }

    // Row sums: reduce over the 16 lanes of each half-wave, then atomics.
#pragma unroll
    for (int r = 0; r < 8; ++r) {
        float nv = numacc[r], dv = denacc[r];
#pragma unroll
        for (int m = 1; m < 16; m <<= 1) {
            nv += __shfl_xor(nv, m, 32);
            dv += __shfl_xor(dv, m, 32);
        }
        if (l15 == 0) {
            const int row = wrow + r + half * 8;
            atomicAdd(&num[row], nv);
            atomicAdd(&den[row], dv);
        }
    }
}

// ---------------------------------------------------------------------------
// Kernel 3: loss = -(1/batch) * sum_i [ log(num_i) - log(den_i) ]
// ---------------------------------------------------------------------------
__global__ __launch_bounds__(256) void simloss_final(const float* __restrict__ num,
                                                     const float* __restrict__ den,
                                                     float* __restrict__ out) {
    float s = 0.f;
    for (int i = threadIdx.x; i < NROWS; i += 256)
        s += logf(num[i]) - logf(den[i]);
#pragma unroll
    for (int m = 1; m < 32; m <<= 1) s += __shfl_xor(s, m, 32);
    __shared__ float red[8];
    if ((threadIdx.x & 31) == 0) red[threadIdx.x >> 5] = s;
    __syncthreads();
    if (threadIdx.x == 0) {
        float t = 0.f;
        for (int w = 0; w < 8; ++w) t += red[w];
        out[0] = -t / (float)BATCH_N;
    }
}

// ---------------------------------------------------------------------------
extern "C" void kernel_launch(void* const* d_in, const int* in_sizes, int n_in,
                              void* d_out, int out_size, void* d_ws, size_t ws_size,
                              hipStream_t stream) {
    const float* feat = (const float*)d_in[0];
    float* out = (float*)d_out;

    _Float16* fnh = (_Float16*)d_ws;
    float* num = (float*)((char*)d_ws + (size_t)NROWS * DIM * sizeof(_Float16));
    float* den = num + NROWS;

    simloss_norm<<<NROWS, 256, 0, stream>>>(feat, fnh, num, den);

    dim3 g2(NROWS / ROWS_PER_BLK, YSPLIT);
    simloss_gemm<<<g2, 256, 0, stream>>>(fnh, num, den);

    simloss_final<<<1, 256, 0, stream>>>(num, den, out);
}
